// MultiHeadSelfAttention_82781199663897
// MI455X (gfx1250) — matrix operand order repaired and checked by reading, not compiled, not run on hardware
//
#include <hip/hip_runtime.h>
#include <math.h>
#include <stdint.h>

typedef __attribute__((ext_vector_type(16))) __bf16          v16bf;
typedef __attribute__((ext_vector_type(8)))  float           v8f;
typedef __attribute__((ext_vector_type(8)))  unsigned short  u16x8;
typedef __attribute__((ext_vector_type(4)))  unsigned short  u16x4;
typedef __attribute__((ext_vector_type(4)))  unsigned int    v4u;
typedef __attribute__((ext_vector_type(8)))  int             v8i;
typedef __attribute__((ext_vector_type(4)))  int             v4i;

union Frag { u16x8 h[2]; v16bf v; };

#define NSEQ 2048
#define BSZ  2
#define EDIM 1024
#define NH   16
#define HD   64
#define NTOK 4096           // NSEQ*BSZ
#define F3   3072           // 3*EDIM
#define LDSW 40             // GEMM LDS row stride (u16): 32 data + 4-DWORD TDM pad
#define KVW  72             // attn K/V/P LDS row stride (u16): 64 data + 4-DWORD pad

static __device__ __forceinline__ unsigned short f2bf(float f) {
    union { float f; unsigned u; } v; v.f = f;
    unsigned r = v.u + 0x7FFFu + ((v.u >> 16) & 1u);   // RNE
    return (unsigned short)(r >> 16);
}

// ---- Tensor Data Mover: 2D bf16 tile -> LDS with per-row padding -------------
// pad_iv code: interval = 2^(code+1) DWORDs; pad_amt code: amount = code+1 DWORDs
static __device__ __forceinline__ void tdm_load_2d(
    const unsigned short* lds_dst, const unsigned short* gsrc,
    unsigned tensor_d0, unsigned tensor_d1,
    unsigned tile_d0, unsigned tile_d1,
    unsigned stride_d0, int pad_iv, int pad_amt)
{
    unsigned long long ga = (unsigned long long)(uintptr_t)gsrc;
    v4u g0;
    g0[0] = 1u;                                    // count=1, user descriptor
    g0[1] = (unsigned)(uintptr_t)lds_dst;          // lds_addr (low 32 bits = LDS offset)
    g0[2] = (unsigned)ga;                          // global_addr[31:0]
    g0[3] = (unsigned)(ga >> 32) | (2u << 30);     // global_addr[56:32] | type=2
    v8i g1;
    g1[0] = (1 << 16) | (1 << 20) | (pad_iv << 22) | (pad_amt << 25); // 2B elems, pad
    g1[1] = (int)(tensor_d0 << 16);                                    // dim0[15:0]
    g1[2] = (int)((tensor_d0 >> 16) | (tensor_d1 << 16));              // dim0[31:16]|dim1[15:0]
    g1[3] = (int)((tensor_d1 >> 16) | (tile_d0 << 16));                // dim1[31:16]|tile0
    g1[4] = (int)tile_d1;                                              // tile1 (tile2=0)
    g1[5] = (int)stride_d0;                                            // dim0 stride lo
    g1[6] = 0; g1[7] = 0;
    v4i z4 = {0, 0, 0, 0};
#if defined(__clang_major__) && (__clang_major__ >= 23)
    v8i z8 = {0, 0, 0, 0, 0, 0, 0, 0};
    __builtin_amdgcn_tensor_load_to_lds(g0, g1, z4, z4, z8, 0);
#else
    __builtin_amdgcn_tensor_load_to_lds(g0, g1, z4, z4, 0);
#endif
}

static __device__ __forceinline__ v16bf ldsAfrag(const unsigned short* base, int row, int hf) {
    const unsigned short* p = base + row * LDSW;
    Frag u;
    u.h[0] = *(const u16x8*)(p + hf * 8);
    u.h[1] = *(const u16x8*)(p + 16 + hf * 8);
    return u.v;
}
static __device__ __forceinline__ v16bf ldsBfrag(const unsigned short* base, int row, int hf) {
    const unsigned short* p = base + row * LDSW + hf * 8;
    Frag u;
    u.h[0] = *(const u16x8*)(p);
    u.h[1] = *(const u16x8*)(p + 16);
    return u.v;
}

// ---------------------------------------------------------------------------
// Kernel 0: fp32 -> bf16 bulk convert (seq, w_qkv, w_out)
// ---------------------------------------------------------------------------
__global__ __launch_bounds__(256) void cvt_bf16(
    const float* __restrict__ src, unsigned short* __restrict__ dst, int n4)
{
    int idx = blockIdx.x * 256 + threadIdx.x;
    if (idx >= n4) return;
    float4 f = *(const float4*)(src + (size_t)idx * 4);
    u16x4 o;
    o[0] = f2bf(f.x); o[1] = f2bf(f.y); o[2] = f2bf(f.z); o[3] = f2bf(f.w);
    *(u16x4*)(dst + (size_t)idx * 4) = o;
}

// ---------------------------------------------------------------------------
// Kernels 1&3 core: 128x128 block tile, 8 waves x (64x32), TDM double-buffered
// ---------------------------------------------------------------------------
#define GEMM_PROLOG(NDIM_ROWS)                                                     \
    __shared__ __align__(16) unsigned short ldsA[2][128 * LDSW];                   \
    __shared__ __align__(16) unsigned short ldsB[2][128 * LDSW];                   \
    const int tid  = threadIdx.x;                                                  \
    const int lane = tid & 31, wave = tid >> 5;                                    \
    const int hf   = lane >> 4, l16 = lane & 15;                                   \
    const int tileM = blockIdx.y * 128;                                            \
    const int tileN = blockIdx.x * 128;                                            \
    const int wm = (wave & 1) * 64, wn = (wave >> 1) * 32;                         \
    v8f acc[4][2];                                                                 \
    _Pragma("unroll") for (int mt = 0; mt < 4; ++mt)                               \
        _Pragma("unroll") for (int nt = 0; nt < 2; ++nt)                           \
            acc[mt][nt] = v8f{0.f,0.f,0.f,0.f,0.f,0.f,0.f,0.f};                    \
    if (wave == 0) {                                                               \
        tdm_load_2d(&ldsA[0][0], A + (size_t)tileM * EDIM, EDIM, NTOK, 32, 128, EDIM, 3, 3); \
        tdm_load_2d(&ldsB[0][0], B + (size_t)tileN * EDIM, EDIM, (NDIM_ROWS), 32, 128, EDIM, 3, 3); \
    }                                                                              \
    for (int it = 0; it < EDIM / 32; ++it) {                                       \
        const int cur = it & 1;                                                    \
        if (wave == 0) {                                                           \
            if (it + 1 < EDIM / 32) {                                              \
                const int kn = (it + 1) * 32;                                      \
                tdm_load_2d(&ldsA[cur ^ 1][0], A + (size_t)tileM * EDIM + kn, EDIM, NTOK, 32, 128, EDIM, 3, 3); \
                tdm_load_2d(&ldsB[cur ^ 1][0], B + (size_t)tileN * EDIM + kn, EDIM, (NDIM_ROWS), 32, 128, EDIM, 3, 3); \
                __builtin_amdgcn_s_wait_tensorcnt(2);                              \
            } else {                                                               \
                __builtin_amdgcn_s_wait_tensorcnt(0);                              \
            }                                                                      \
        }                                                                          \
        __syncthreads();                                                           \
        v16bf aF[4], bF[2];                                                        \
        _Pragma("unroll") for (int mt = 0; mt < 4; ++mt)                           \
            aF[mt] = ldsAfrag(&ldsA[cur][0], wm + mt * 16 + l16, hf);              \
        _Pragma("unroll") for (int nt = 0; nt < 2; ++nt)                           \
            bF[nt] = ldsBfrag(&ldsB[cur][0], wn + nt * 16 + l16, hf);              \
        _Pragma("unroll") for (int mt = 0; mt < 4; ++mt)                           \
            _Pragma("unroll") for (int nt = 0; nt < 2; ++nt)                       \
                acc[mt][nt] = __builtin_amdgcn_wmma_f32_16x16x32_bf16(             \
                    false, aF[mt], false, bF[nt], (short)0, acc[mt][nt], false, false); \
        __syncthreads();                                                           \
    }

__global__ __launch_bounds__(256) void qkv_gemm(
    const unsigned short* __restrict__ A, const unsigned short* __restrict__ B,
    const float* __restrict__ bq,
    unsigned short* __restrict__ qb, unsigned short* __restrict__ kb,
    unsigned short* __restrict__ vb)
{
    GEMM_PROLOG(F3)

#pragma unroll
    for (int mt = 0; mt < 4; ++mt) {
#pragma unroll
        for (int nt = 0; nt < 2; ++nt) {
            int f = tileN + wn + nt * 16 + l16;
            float bias = bq[f];
            int sec = f >> 10;            // 0=q,1=k,2=v (uniform per 16-col group)
            int fe  = f & 1023;
            int h   = fe >> 6, d = fe & 63;
#pragma unroll
            for (int vv = 0; vv < 8; ++vv) {
                int t = tileM + wm + mt * 16 + vv + 8 * hf;
                float val = acc[mt][nt][vv] + bias;
                int n = t >> 1, b = t & 1;
                int bh = b * NH + h;
                if (sec == 0)
                    qb[((size_t)bh * NSEQ + n) * HD + d] = f2bf(val * 0.125f);
                else if (sec == 1)
                    kb[((size_t)bh * NSEQ + n) * HD + d] = f2bf(val);
                else
                    vb[((size_t)bh * HD + d) * NSEQ + n] = f2bf(val);
            }
        }
    }
}

__global__ __launch_bounds__(256) void out_gemm(
    const unsigned short* __restrict__ A, const unsigned short* __restrict__ B,
    const float* __restrict__ bias, float* __restrict__ out)
{
    GEMM_PROLOG(EDIM)

#pragma unroll
    for (int mt = 0; mt < 4; ++mt)
#pragma unroll
        for (int nt = 0; nt < 2; ++nt) {
            int f = tileN + wn + nt * 16 + l16;
            float bs = bias[f];
#pragma unroll
            for (int vv = 0; vv < 8; ++vv) {
                int t = tileM + wm + mt * 16 + vv + 8 * hf;
                out[(size_t)t * EDIM + f] = acc[mt][nt][vv] + bs;
            }
        }
}

// ---------------------------------------------------------------------------
// Kernel 2: flash attention per (b,h). Block = 8 waves x 16 q-rows = 128 rows.
// K/V 64-key tiles staged in LDS by TDM (double-buffered), softmax in registers.
// ---------------------------------------------------------------------------
__global__ __launch_bounds__(256) void attn(
    const unsigned short* __restrict__ q, const unsigned short* __restrict__ k,
    const unsigned short* __restrict__ v, unsigned short* __restrict__ o)
{
    __shared__ __align__(16) unsigned short ldsK[2][64 * KVW];
    __shared__ __align__(16) unsigned short ldsV[2][64 * KVW];
    __shared__ __align__(16) unsigned short sP[8][16 * KVW];

    const int tid  = threadIdx.x;
    const int lane = tid & 31, wave = tid >> 5;
    const int hf   = lane >> 4, l16 = lane & 15;
    const int bh = blockIdx.y;
    const int b  = bh >> 4, h = bh & 15;

    const unsigned short* Qb = q + (size_t)bh * NSEQ * HD;
    const unsigned short* Kb = k + (size_t)bh * NSEQ * HD;
    const unsigned short* Vt = v + (size_t)bh * HD * NSEQ;
    const int q0 = blockIdx.x * 128 + wave * 16;

    // Q fragments (A-operand), loaded once: row = q-token, K-dim = d
    v16bf aQ[2];
#pragma unroll
    for (int ds = 0; ds < 2; ++ds) {
        const unsigned short* p = Qb + (size_t)(q0 + l16) * HD + ds * 32 + hf * 8;
        Frag u; u.h[0] = *(const u16x8*)p; u.h[1] = *(const u16x8*)(p + 16);
        aQ[ds] = u.v;
    }

    float mrow[8], lrow[8];
#pragma unroll
    for (int vv = 0; vv < 8; ++vv) { mrow[vv] = -INFINITY; lrow[vv] = 0.f; }
    v8f oAcc[4];
#pragma unroll
    for (int nt = 0; nt < 4; ++nt) oAcc[nt] = v8f{0.f,0.f,0.f,0.f,0.f,0.f,0.f,0.f};

    if (wave == 0) {
        // K tile: rows=keys (stride HD), tile 64(d) x 64(keys); V^T tile: rows=d (stride NSEQ)
        tdm_load_2d(&ldsK[0][0], Kb, HD, NSEQ, 64, 64, HD, 4, 3);
        tdm_load_2d(&ldsV[0][0], Vt, NSEQ, HD, 64, 64, NSEQ, 4, 3);
    }

    for (int it = 0; it < NSEQ / 64; ++it) {
        const int cur = it & 1;
        if (wave == 0) {
            if (it + 1 < NSEQ / 64) {
                const int kn = (it + 1) * 64;
                tdm_load_2d(&ldsK[cur ^ 1][0], Kb + (size_t)kn * HD, HD, NSEQ, 64, 64, HD, 4, 3);
                tdm_load_2d(&ldsV[cur ^ 1][0], Vt + kn, NSEQ, HD, 64, 64, NSEQ, 4, 3);
                __builtin_amdgcn_s_wait_tensorcnt(2);
            } else {
                __builtin_amdgcn_s_wait_tensorcnt(0);
            }
        }
        __syncthreads();
        const unsigned short* Kt = &ldsK[cur][0];
        const unsigned short* Vtile = &ldsV[cur][0];

        // S = Q * K^T  (16 x 64)
        v8f sAcc[4];
#pragma unroll
        for (int nt = 0; nt < 4; ++nt) sAcc[nt] = v8f{0.f,0.f,0.f,0.f,0.f,0.f,0.f,0.f};
#pragma unroll
        for (int nt = 0; nt < 4; ++nt) {
#pragma unroll
            for (int ds = 0; ds < 2; ++ds) {
                const unsigned short* p = &Kt[(nt * 16 + l16) * KVW + ds * 32 + hf * 8];
                Frag u; u.h[0] = *(const u16x8*)p; u.h[1] = *(const u16x8*)(p + 16);
                sAcc[nt] = __builtin_amdgcn_wmma_f32_16x16x32_bf16(
                    false, aQ[ds], false, u.v, (short)0, sAcc[nt], false, false);
            }
        }

        // online softmax in registers: row r = vv + 8*hf, reduce across 16 lanes
        float al[8];
#pragma unroll
        for (int vv = 0; vv < 8; ++vv) {
            float x = fmaxf(fmaxf(sAcc[0][vv], sAcc[1][vv]), fmaxf(sAcc[2][vv], sAcc[3][vv]));
            x = fmaxf(x, __shfl_xor(x, 1, 32));
            x = fmaxf(x, __shfl_xor(x, 2, 32));
            x = fmaxf(x, __shfl_xor(x, 4, 32));
            x = fmaxf(x, __shfl_xor(x, 8, 32));
            float mnew = fmaxf(mrow[vv], x);
            al[vv] = __expf(mrow[vv] - mnew);
            mrow[vv] = mnew;
        }
#pragma unroll
        for (int vv = 0; vv < 8; ++vv) {
            float rs = 0.f;
            int prow = (vv + 8 * hf) * KVW + l16;
#pragma unroll
            for (int nt = 0; nt < 4; ++nt) {
                float pe = __expf(sAcc[nt][vv] - mrow[vv]);
                rs += pe;
                sP[wave][prow + nt * 16] = f2bf(pe);
            }
            rs += __shfl_xor(rs, 1, 32);
            rs += __shfl_xor(rs, 2, 32);
            rs += __shfl_xor(rs, 4, 32);
            rs += __shfl_xor(rs, 8, 32);
            lrow[vv] = lrow[vv] * al[vv] + rs;
        }
#pragma unroll
        for (int nt = 0; nt < 4; ++nt)
#pragma unroll
            for (int vv = 0; vv < 8; ++vv)
                oAcc[nt][vv] *= al[vv];
        asm volatile("" ::: "memory");

        // O += P * V  (A = P from LDS transpose, B = V^T tile)
        v16bf aP[2];
#pragma unroll
        for (int ks = 0; ks < 2; ++ks) {
            const unsigned short* p = &sP[wave][l16 * KVW + ks * 32 + hf * 8];
            Frag u; u.h[0] = *(const u16x8*)p; u.h[1] = *(const u16x8*)(p + 16);
            aP[ks] = u.v;
        }
#pragma unroll
        for (int nt = 0; nt < 4; ++nt) {
#pragma unroll
            for (int ks = 0; ks < 2; ++ks) {
                const unsigned short* p = &Vtile[(nt * 16 + l16) * KVW + ks * 32 + hf * 8];
                Frag u; u.h[0] = *(const u16x8*)p; u.h[1] = *(const u16x8*)(p + 16);
                oAcc[nt] = __builtin_amdgcn_wmma_f32_16x16x32_bf16(
                    false, aP[ks], false, u.v, (short)0, oAcc[nt], false, false);
            }
        }
        __syncthreads();
    }

    // normalize and store O as bf16 [t][e]
    float linv[8];
#pragma unroll
    for (int vv = 0; vv < 8; ++vv) linv[vv] = 1.0f / lrow[vv];
#pragma unroll
    for (int nt = 0; nt < 4; ++nt)
#pragma unroll
        for (int vv = 0; vv < 8; ++vv) {
            int n = q0 + vv + 8 * hf;
            int t = n * BSZ + b;
            int col = h * HD + nt * 16 + l16;
            o[(size_t)t * EDIM + col] = f2bf(oAcc[nt][vv] * linv[vv]);
        }
}

// ---------------------------------------------------------------------------
extern "C" void kernel_launch(void* const* d_in, const int* in_sizes, int n_in,
                              void* d_out, int out_size, void* d_ws, size_t ws_size,
                              hipStream_t stream)
{
    const float* seq   = (const float*)d_in[0];   // (2048, 2, 1024)
    const float* w_qkv = (const float*)d_in[1];   // (3072, 1024)
    const float* b_qkv = (const float*)d_in[2];   // (3072,)
    const float* w_out = (const float*)d_in[3];   // (1024, 1024)
    const float* b_out = (const float*)d_in[4];   // (1024,)
    float* out = (float*)d_out;                   // (2048, 2, 1024) fp32

    // workspace (all bf16/u16): Q,K,V,O + converted X, Wqkv, Wout = 48 MB
    const size_t per = (size_t)BSZ * NH * NSEQ * HD;   // 4,194,304
    unsigned short* qb  = (unsigned short*)d_ws;
    unsigned short* kb  = qb + per;
    unsigned short* vb  = kb + per;
    unsigned short* ob  = vb + per;
    unsigned short* xbf = ob + per;
    unsigned short* wqb = xbf + (size_t)NTOK * EDIM;
    unsigned short* wob = wqb + (size_t)F3 * EDIM;

    cvt_bf16<<<dim3((NTOK * EDIM / 4) / 256), 256, 0, stream>>>(seq, xbf, NTOK * EDIM / 4);
    cvt_bf16<<<dim3((F3 * EDIM / 4) / 256), 256, 0, stream>>>(w_qkv, wqb, F3 * EDIM / 4);
    cvt_bf16<<<dim3((EDIM * EDIM / 4) / 256), 256, 0, stream>>>(w_out, wob, EDIM * EDIM / 4);

    qkv_gemm<<<dim3(F3 / 128, NTOK / 128), 256, 0, stream>>>(xbf, wqb, b_qkv, qb, kb, vb);
    attn<<<dim3(NSEQ / 128, BSZ * NH), 256, 0, stream>>>(qb, kb, vb, ob);
    out_gemm<<<dim3(EDIM / 128, NTOK / 128), 256, 0, stream>>>(ob, wob, b_out, out);
}
